// HeteroIncidentGATv2_57105885167843
// MI455X (gfx1250) — compile-verified, matrix operand order checked by reading, not run on hardware
//
#include <hip/hip_runtime.h>
#include <hip/hip_bf16.h>
#include <stdint.h>

// ---------------------------------------------------------------------------
// HeteroIncidentGATv2 for MI455X (gfx1250, wave32, WMMA)
// ---------------------------------------------------------------------------

typedef __attribute__((ext_vector_type(16))) _Float16 v16h;
typedef __attribute__((ext_vector_type(8)))  float    v8f;

#define NHOST   50000
#define NVM     100000
#define NJOB    200000
#define NSWITCH 5000
#define EHS     200000
#define EVH     400000
#define EJV     800000
#define HC      128
#define INDIM   64

union H4 { uint2  u; _Float16 h[4]; };
union F4 { float4 v; float    f[4]; };

// -------------------------------- utilities --------------------------------

__global__ void fill_u32_kernel(uint32_t* __restrict__ p, uint32_t v, long n) {
    long i = (long)blockIdx.x * blockDim.x + threadIdx.x;
    if (i < n) p[i] = v;
}

// W: [64][128] row major f32  ->  WT: [128][64] f16 (n-major, so each WMMA B
// fragment lane reads one output column's K-run contiguously)
__global__ void convert_w_kernel(const float* __restrict__ W, _Float16* __restrict__ WT) {
    int i = blockIdx.x * blockDim.x + threadIdx.x;
    if (i < INDIM * HC) {
        int k = i >> 7, n = i & 127;
        WT[n * INDIM + k] = (_Float16)W[i];
    }
}

__device__ __forceinline__ void atomicMaxF(float* addr, float v) {
    if (v >= 0.f) atomicMax((int*)addr, __float_as_int(v));
    else          atomicMin((unsigned int*)addr, (unsigned int)__float_as_int(v));
}

// ------------------------------ WMMA node GEMM ------------------------------
// Y[M,128] = f16( X[M,64] @ W[64,128] + bias ), stored f16.
// Block: 128 threads = 4 waves; each block does a 16-row x 128-col tile.
// Wave w owns columns [w*32, w*32+32): two 16x16 WMMA N-tiles, K=64 in two
// k=32 steps -> 4 v_wmma_f32_16x16x32_f16 per wave. Epilogue stages the tile
// in LDS so global stores are fully coalesced b128s (one uniform tail branch).
__global__ void __launch_bounds__(128)
gemm_node_kernel(const float* __restrict__ X, const _Float16* __restrict__ WT,
                 const float* __restrict__ bias, _Float16* __restrict__ Y, int M)
{
    __shared__ _Float16 As[16 * INDIM];    // 2 KB
    __shared__ _Float16 Bs[HC * INDIM];    // 16 KB
    __shared__ _Float16 Cs[16 * HC];       // 4 KB staging for coalesced stores

    const int tid = threadIdx.x;
    const int m0  = blockIdx.x * 16;

    // Stage A tile (f32 -> f16), zero-pad out-of-range rows.
    for (int i = tid; i < 16 * INDIM; i += 128) {
        int r = i >> 6, c = i & 63;
        int row = m0 + r;
        float v = (row < M) ? X[(size_t)row * INDIM + c] : 0.f;
        As[i] = (_Float16)v;
    }
    // Stage all 128x64 f16 weights (dword copies -> b128-friendly).
    {
        const uint32_t* src = (const uint32_t*)WT;
        uint32_t*       dst = (uint32_t*)Bs;
        for (int i = tid; i < (HC * INDIM) / 2; i += 128) dst[i] = src[i];
    }
    __syncthreads();

    const int lane = tid & 31;
    const int wave = tid >> 5;
    const int mrow = lane & 15;   // A fragment: row id
    const int grp  = lane >> 4;   // lane group selects K sub-range

    #pragma unroll
    for (int nt = 0; nt < 2; ++nt) {
        const int ncol = (wave * 2 + nt) * 16 + (lane & 15);
        v8f acc = {};
        #pragma unroll
        for (int ks = 0; ks < 2; ++ks) {
            const int k0 = ks * 32;
            v16h a, b;
            #pragma unroll
            for (int j = 0; j < 8; ++j) {
                // 16-bit A 16x32 fragment K mapping (ISA 7.12.2):
                // vgpr j holds K = (j>=4?16:0) + grp*8 + (j&3)*2 , +1
                const int kk = k0 + ((j & 4) << 2) + grp * 8 + ((j & 3) << 1);
                a[2 * j]     = As[mrow * INDIM + kk];
                a[2 * j + 1] = As[mrow * INDIM + kk + 1];
                b[2 * j]     = Bs[ncol * INDIM + kk];
                b[2 * j + 1] = Bs[ncol * INDIM + kk + 1];
            }
            acc = __builtin_amdgcn_wmma_f32_16x16x32_f16(
                false, a, false, b, (short)0, acc, false, false);
        }
        const float bn = bias[ncol];
        #pragma unroll
        for (int r = 0; r < 8; ++r)        // C/D layout: vgpr r -> M = r + 8*grp
            Cs[(r + 8 * grp) * HC + ncol] = (_Float16)(acc[r] + bn);
    }
    __syncthreads();

    // Coalesced writeback: tile is 16 rows x 128 f16 = 1024 dwords contiguous.
    if (m0 + 16 <= M) {                    // uniform fast path (all but last block)
        const uint4* s4 = (const uint4*)Cs;
        uint4*       d4 = (uint4*)(Y + (size_t)m0 * HC);
        d4[tid]       = s4[tid];
        d4[tid + 128] = s4[tid + 128];
    } else {
        const uint32_t* s32 = (const uint32_t*)Cs;
        uint32_t*       y32 = (uint32_t*)Y;
        for (int i = tid; i < 16 * (HC / 2); i += 128) {
            int row = m0 + (i >> 6);       // 64 dwords per row
            if (row < M) y32[(size_t)m0 * (HC / 2) + i] = s32[i];
        }
    }
}

// ------------------------- edge pass 1: logits + max ------------------------
// One wave per edge; lane owns 4 contiguous channels (one b64 gather per row),
// head = lane>>3; segmented shuffle reduction over 8-lane groups.
__global__ void __launch_bounds__(256)
edge_logits_kernel(const int* __restrict__ src, const int* __restrict__ dst,
                   const _Float16* __restrict__ xl, const _Float16* __restrict__ xr,
                   const float* __restrict__ att, float* __restrict__ mbuf,
                   float* __restrict__ logit_out, int E)
{
    const int e    = blockIdx.x * 8 + (threadIdx.x >> 5);
    const int lane = threadIdx.x & 31;
    if (e >= E) return;
    const int s = src[e], d = dst[e];
    const int head = lane >> 3;

    H4 va, vb; F4 at;
    va.u = *(const uint2*)(xl + (size_t)s * HC + 4 * lane);
    vb.u = *(const uint2*)(xr + (size_t)d * HC + 4 * lane);
    at.v = *(const float4*)(att + 4 * lane);

    float v = 0.f;
    #pragma unroll
    for (int j = 0; j < 4; ++j) {
        float t = (float)va.h[j] + (float)vb.h[j];
        t = (t > 0.f) ? t : 0.2f * t;                 // GATv2 leaky slope
        v += t * at.f[j];
    }
    v += __shfl_xor(v, 1, 32);
    v += __shfl_xor(v, 2, 32);
    v += __shfl_xor(v, 4, 32);
    if ((lane & 7) == 0) {
        logit_out[(size_t)e * 4 + head] = v;
        atomicMaxF(&mbuf[d * 4 + head], v);
    }
}

// --------------- edge pass 2: exp, denom, weighted aggregation --------------
// exbuf holds logits on entry, exp(logit - max) on exit (in-place in d_out's
// alpha region). accum gets sum(xl[src] * ex) via f32 atomics (L2-resident).
__global__ void __launch_bounds__(256)
edge_accum_kernel(const int* __restrict__ src, const int* __restrict__ dst,
                  const _Float16* __restrict__ xl, const float* __restrict__ mbuf,
                  float* __restrict__ exbuf, float* __restrict__ denom,
                  float* __restrict__ accum, int E)
{
    const int e    = blockIdx.x * 8 + (threadIdx.x >> 5);
    const int lane = threadIdx.x & 31;
    if (e >= E) return;
    const int s = src[e], d = dst[e];
    const int head = lane >> 3;

    const float l  = exbuf[(size_t)e * 4 + head];     // broadcast within group
    const float ex = __expf(l - mbuf[d * 4 + head]);
    if ((lane & 7) == 0) {
        exbuf[(size_t)e * 4 + head] = ex;
        atomicAdd(&denom[d * 4 + head], ex);
    }
    H4 va;
    va.u = *(const uint2*)(xl + (size_t)s * HC + 4 * lane);
    #pragma unroll
    for (int j = 0; j < 4; ++j)
        atomicAdd(&accum[(size_t)d * HC + 4 * lane + j], (float)va.h[j] * ex);
}

// ---------------- edge pass 3: alpha = ex / denom[dst] (in place) -----------
// One thread per edge: one dst lookup, one float4 in/out.
__global__ void alpha_norm_kernel(const int* __restrict__ dst,
                                  const float* __restrict__ denom,
                                  float* __restrict__ alpha, int E)
{
    const int e = blockIdx.x * blockDim.x + threadIdx.x;
    if (e < E) {
        const int d = dst[e];
        F4 dn, a;
        dn.v = *(const float4*)(denom + (size_t)d * 4);
        a.v  = *(const float4*)(alpha + (size_t)e * 4);
        #pragma unroll
        for (int j = 0; j < 4; ++j) a.f[j] /= dn.f[j];
        *(float4*)(alpha + (size_t)e * 4) = a.v;
    }
}

// ------------- node epilogue: divide, +bias, leaky, dot, sigmoid ------------
__global__ void __launch_bounds__(256)
node_out_kernel(const float* __restrict__ accum, const float* __restrict__ denom,
                const float* __restrict__ bias, const float* __restrict__ Wnt,
                const float* __restrict__ bnt, float* __restrict__ score, int N)
{
    const int n    = blockIdx.x * 8 + (threadIdx.x >> 5);
    const int lane = threadIdx.x & 31;
    if (n >= N) return;
    const int head = lane >> 3;

    const float dn  = denom[n * 4 + head];
    const float inv = (dn > 0.f) ? 1.f / dn : 0.f;    // empty segment -> 0
    F4 ac, bi, wn;
    ac.v = *(const float4*)(accum + (size_t)n * HC + 4 * lane);
    bi.v = *(const float4*)(bias + 4 * lane);
    wn.v = *(const float4*)(Wnt + 4 * lane);

    float dot = 0.f;
    #pragma unroll
    for (int j = 0; j < 4; ++j) {
        float v = ac.f[j] * inv + bi.f[j];
        v = (v > 0.f) ? v : 0.01f * v;                // output leaky slope
        dot += v * wn.f[j];
    }
    #pragma unroll
    for (int off = 16; off > 0; off >>= 1) dot += __shfl_xor(dot, off, 32);
    if (lane == 0) score[n] = 1.f / (1.f + __expf(-(dot + bnt[0])));
}

// job nodes: h = 0 -> score = sigmoid(b_job)
__global__ void job_score_kernel(const float* __restrict__ b, float* __restrict__ score, int N) {
    const int i = blockIdx.x * blockDim.x + threadIdx.x;
    if (i < N) score[i] = 1.f / (1.f + __expf(-b[0]));
}

// --------------------------------- launch -----------------------------------

extern "C" void kernel_launch(void* const* d_in, const int* in_sizes, int n_in,
                              void* d_out, int out_size, void* d_ws, size_t ws_size,
                              hipStream_t stream)
{
    // ---- inputs (setup_inputs dict order) ----
    const float* x_host   = (const float*)d_in[0];
    const float* x_vm     = (const float*)d_in[1];
    const float* x_job    = (const float*)d_in[2];
    const float* x_switch = (const float*)d_in[3];
    const int* e_hs_src = (const int*)d_in[4];
    const int* e_hs_dst = (const int*)d_in[5];
    const int* e_vh_src = (const int*)d_in[6];
    const int* e_vh_dst = (const int*)d_in[7];
    const int* e_jv_src = (const int*)d_in[8];
    const int* e_jv_dst = (const int*)d_in[9];
    // per relation: Wl, bl, Wr, br, att, bias
    const float* Wl_hs = (const float*)d_in[10]; const float* bl_hs = (const float*)d_in[11];
    const float* Wr_hs = (const float*)d_in[12]; const float* br_hs = (const float*)d_in[13];
    const float* att_hs = (const float*)d_in[14]; const float* bias_hs = (const float*)d_in[15];
    const float* Wl_vh = (const float*)d_in[16]; const float* bl_vh = (const float*)d_in[17];
    const float* Wr_vh = (const float*)d_in[18]; const float* br_vh = (const float*)d_in[19];
    const float* att_vh = (const float*)d_in[20]; const float* bias_vh = (const float*)d_in[21];
    const float* Wl_jv = (const float*)d_in[22]; const float* bl_jv = (const float*)d_in[23];
    const float* Wr_jv = (const float*)d_in[24]; const float* br_jv = (const float*)d_in[25];
    const float* att_jv = (const float*)d_in[26]; const float* bias_jv = (const float*)d_in[27];
    const float* W_host = (const float*)d_in[28]; const float* b_host = (const float*)d_in[29];
    const float* W_vm   = (const float*)d_in[30]; const float* b_vm   = (const float*)d_in[31];
    const float* W_job  = (const float*)d_in[32]; const float* b_job  = (const float*)d_in[33];
    const float* W_sw   = (const float*)d_in[34]; const float* b_sw   = (const float*)d_in[35];

    // ---- outputs: [scores 355000][a_hs 800000][a_vh 1600000][a_jv 3200000] ----
    float* out        = (float*)d_out;
    float* s_host     = out;
    float* s_vm       = out + NHOST;
    float* s_job      = out + NHOST + NVM;
    float* s_switch   = out + NHOST + NVM + NJOB;
    float* a_hs       = out + (NHOST + NVM + NJOB + NSWITCH);
    float* a_vh       = a_hs + (size_t)EHS * 4;
    float* a_jv       = a_vh + (size_t)EVH * 4;

    // ---- workspace carve ----
    char* w = (char*)d_ws;
    auto take = [&](size_t bytes) { void* p = (void*)w; w += bytes; return p; };
    _Float16* WTl_hs = (_Float16*)take((size_t)HC * INDIM * 2);
    _Float16* WTr_hs = (_Float16*)take((size_t)HC * INDIM * 2);
    _Float16* WTl_vh = (_Float16*)take((size_t)HC * INDIM * 2);
    _Float16* WTr_vh = (_Float16*)take((size_t)HC * INDIM * 2);
    _Float16* WTl_jv = (_Float16*)take((size_t)HC * INDIM * 2);
    _Float16* WTr_jv = (_Float16*)take((size_t)HC * INDIM * 2);
    _Float16* xl_hs = (_Float16*)take((size_t)NHOST   * HC * 2);
    _Float16* xr_hs = (_Float16*)take((size_t)NSWITCH * HC * 2);
    _Float16* xl_vh = (_Float16*)take((size_t)NVM     * HC * 2);
    _Float16* xr_vh = (_Float16*)take((size_t)NHOST   * HC * 2);
    _Float16* xl_jv = (_Float16*)take((size_t)NJOB    * HC * 2);
    _Float16* xr_jv = (_Float16*)take((size_t)NVM     * HC * 2);
    // contiguous groups for bulk fills:
    float* m_sw   = (float*)take((size_t)NSWITCH * 4 * 4);
    float* m_host = (float*)take((size_t)NHOST   * 4 * 4);
    float* m_vm   = (float*)take((size_t)NVM     * 4 * 4);
    float* den_sw   = (float*)take((size_t)NSWITCH * 4 * 4);
    float* den_host = (float*)take((size_t)NHOST   * 4 * 4);
    float* den_vm   = (float*)take((size_t)NVM     * 4 * 4);
    float* acc_sw   = (float*)take((size_t)NSWITCH * HC * 4);
    float* acc_host = (float*)take((size_t)NHOST   * HC * 4);
    float* acc_vm   = (float*)take((size_t)NVM     * HC * 4);
    (void)ws_size; (void)n_in; (void)in_sizes; (void)out_size;

    const int NDST4  = (NSWITCH + NHOST + NVM) * 4;           // 620000
    const long ZCNT  = (long)NDST4 + (long)(NSWITCH + NHOST + NVM) * HC; // den+acc

    // ---- init ----
    fill_u32_kernel<<<(NDST4 + 255) / 256, 256, 0, stream>>>(
        (uint32_t*)m_sw, 0xFF800000u, NDST4);                 // -inf
    fill_u32_kernel<<<(int)((ZCNT + 255) / 256), 256, 0, stream>>>(
        (uint32_t*)den_sw, 0u, ZCNT);                         // denom + accum

    // ---- weight convert (f32 [64][128] -> f16 [128][64]) ----
    convert_w_kernel<<<(INDIM * HC + 255) / 256, 256, 0, stream>>>(Wl_hs, WTl_hs);
    convert_w_kernel<<<(INDIM * HC + 255) / 256, 256, 0, stream>>>(Wr_hs, WTr_hs);
    convert_w_kernel<<<(INDIM * HC + 255) / 256, 256, 0, stream>>>(Wl_vh, WTl_vh);
    convert_w_kernel<<<(INDIM * HC + 255) / 256, 256, 0, stream>>>(Wr_vh, WTr_vh);
    convert_w_kernel<<<(INDIM * HC + 255) / 256, 256, 0, stream>>>(Wl_jv, WTl_jv);
    convert_w_kernel<<<(INDIM * HC + 255) / 256, 256, 0, stream>>>(Wr_jv, WTr_jv);

    // ---- WMMA node GEMMs ----
    gemm_node_kernel<<<(NHOST   + 15) / 16, 128, 0, stream>>>(x_host,   WTl_hs, bl_hs, xl_hs, NHOST);
    gemm_node_kernel<<<(NSWITCH + 15) / 16, 128, 0, stream>>>(x_switch, WTr_hs, br_hs, xr_hs, NSWITCH);
    gemm_node_kernel<<<(NVM     + 15) / 16, 128, 0, stream>>>(x_vm,     WTl_vh, bl_vh, xl_vh, NVM);
    gemm_node_kernel<<<(NHOST   + 15) / 16, 128, 0, stream>>>(x_host,   WTr_vh, br_vh, xr_vh, NHOST);
    gemm_node_kernel<<<(NJOB    + 15) / 16, 128, 0, stream>>>(x_job,    WTl_jv, bl_jv, xl_jv, NJOB);
    gemm_node_kernel<<<(NVM     + 15) / 16, 128, 0, stream>>>(x_vm,     WTr_jv, br_jv, xr_jv, NVM);

    // ---- edge pass 1: logits + segment max ----
    edge_logits_kernel<<<(EHS + 7) / 8, 256, 0, stream>>>(e_hs_src, e_hs_dst, xl_hs, xr_hs, att_hs, m_sw,   a_hs, EHS);
    edge_logits_kernel<<<(EVH + 7) / 8, 256, 0, stream>>>(e_vh_src, e_vh_dst, xl_vh, xr_vh, att_vh, m_host, a_vh, EVH);
    edge_logits_kernel<<<(EJV + 7) / 8, 256, 0, stream>>>(e_jv_src, e_jv_dst, xl_jv, xr_jv, att_jv, m_vm,   a_jv, EJV);

    // ---- edge pass 2: exp + denom + weighted aggregation ----
    edge_accum_kernel<<<(EHS + 7) / 8, 256, 0, stream>>>(e_hs_src, e_hs_dst, xl_hs, m_sw,   a_hs, den_sw,   acc_sw,   EHS);
    edge_accum_kernel<<<(EVH + 7) / 8, 256, 0, stream>>>(e_vh_src, e_vh_dst, xl_vh, m_host, a_vh, den_host, acc_host, EVH);
    edge_accum_kernel<<<(EJV + 7) / 8, 256, 0, stream>>>(e_jv_src, e_jv_dst, xl_jv, m_vm,   a_jv, den_vm,   acc_vm,   EJV);

    // ---- edge pass 3: alpha normalize (in place in d_out) ----
    alpha_norm_kernel<<<(EHS + 255) / 256, 256, 0, stream>>>(e_hs_dst, den_sw,   a_hs, EHS);
    alpha_norm_kernel<<<(EVH + 255) / 256, 256, 0, stream>>>(e_vh_dst, den_host, a_vh, EVH);
    alpha_norm_kernel<<<(EJV + 255) / 256, 256, 0, stream>>>(e_jv_dst, den_vm,   a_jv, EJV);

    // ---- node epilogue + scores ----
    node_out_kernel<<<(NHOST   + 7) / 8, 256, 0, stream>>>(acc_host, den_host, bias_vh, W_host, b_host, s_host,   NHOST);
    node_out_kernel<<<(NVM     + 7) / 8, 256, 0, stream>>>(acc_vm,   den_vm,   bias_jv, W_vm,   b_vm,   s_vm,     NVM);
    node_out_kernel<<<(NSWITCH + 7) / 8, 256, 0, stream>>>(acc_sw,   den_sw,   bias_hs, W_sw,   b_sw,   s_switch, NSWITCH);
    job_score_kernel<<<(NJOB + 255) / 256, 256, 0, stream>>>(b_job, s_job, NJOB);
    (void)W_job;
}
